// GINEncoder_71734543777909
// MI455X (gfx1250) — matrix operand order repaired and checked by reading, not compiled, not run on hardware
//
#include <hip/hip_runtime.h>

typedef _Float16 v16h __attribute__((ext_vector_type(16)));
typedef _Float16 v8h  __attribute__((ext_vector_type(8)));
typedef float    v8f  __attribute__((ext_vector_type(8)));
typedef float    f32x4 __attribute__((ext_vector_type(4)));

#define BN_EPS 1e-5f

// ---------------------------------------------------------------------------
// WMMA GEMM: C[nrows x NC] = A[nrows x K] @ W[K x NC], fused epilogue.
// EPI: 0 = +bias -> f32 out ; 1 = BN+ReLU -> f16 out ; 2 = BN+ReLU -> f32 out
// One 256-thread block = 8 waves; each wave owns a 16-row slab and loops all
// 16-wide column tiles. Weights staged transposed (f16) in LDS; epilogue
// constants precomputed per-lane before the barrier; uniform full-tile fast
// path for stores.
// ---------------------------------------------------------------------------
template<int K, int NC, int EPI, bool A_HALF>
__global__ __launch_bounds__(256) void gemm_wmma(
    const void* __restrict__ Av, const float* __restrict__ W,
    const float* __restrict__ pb, const float* __restrict__ pg,
    const float* __restrict__ pbe, const float* __restrict__ pm,
    const float* __restrict__ pv,
    float* __restrict__ outF, _Float16* __restrict__ outH, int nrows)
{
    constexpr int NCT = NC / 16;              // column tiles
    constexpr int KS  = K / 32;               // k-steps

    __shared__ _Float16 wt[NC * K];           // wt[n*K + k] = (f16) W[k*NC + n]

    const int lane = threadIdx.x & 31;
    const int wid  = threadIdx.x >> 5;
    const int row_base = blockIdx.x * 128 + wid * 16;
    const int m  = lane & 15;                 // A: row / B,C,D: column within tile
    const int hi = lane >> 4;                 // half-wave select
    const int kb = hi * 8;                    // A K-offset within 16-group

    // ---- epilogue constants (per lane, per column tile): overlap with staging
    float sArr[NCT], tArr[NCT];
    #pragma unroll
    for (int c = 0; c < NCT; ++c) {
        const int col = c * 16 + m;
        if constexpr (EPI == 0) { sArr[c] = 1.0f; tArr[c] = pb[col]; }
        else {
            float inv = rsqrtf(pv[col] + BN_EPS);
            sArr[c] = pg[col] * inv;
            tArr[c] = (pb[col] - pm[col]) * sArr[c] + pbe[col];  // bias folded in
        }
    }

    // ---- stage W transposed into LDS as f16
    for (int idx = threadIdx.x; idx < NC * K; idx += 256) {
        int n = idx / K, k = idx % K;
        wt[idx] = (_Float16)W[k * NC + n];
    }
    __syncthreads();

    // ---- A fragments for the whole K range, kept in registers
    long arow = row_base + m;
    if (arow >= nrows) arow = nrows - 1;      // clamp loads; WMMA needs EXEC all-1

    v16h afrag[KS];
    #pragma unroll
    for (int ks = 0; ks < KS; ++ks) {
        const int k0 = ks * 32;
        if constexpr (A_HALF) {
            const _Float16* ap = (const _Float16*)Av + arow * K;
            v8h g0 = *(const v8h*)(ap + k0 + kb);
            v8h g1 = *(const v8h*)(ap + k0 + 16 + kb);
            #pragma unroll
            for (int j = 0; j < 8; ++j) { afrag[ks][j] = g0[j]; afrag[ks][8 + j] = g1[j]; }
        } else {
            const float* ap = (const float*)Av + arow * K;
            f32x4 g0 = *(const f32x4*)(ap + k0 + kb);
            f32x4 g1 = *(const f32x4*)(ap + k0 + kb + 4);
            f32x4 g2 = *(const f32x4*)(ap + k0 + 16 + kb);
            f32x4 g3 = *(const f32x4*)(ap + k0 + 16 + kb + 4);
            #pragma unroll
            for (int j = 0; j < 4; ++j) {
                afrag[ks][j]      = (_Float16)g0[j];
                afrag[ks][4 + j]  = (_Float16)g1[j];
                afrag[ks][8 + j]  = (_Float16)g2[j];
                afrag[ks][12 + j] = (_Float16)g3[j];
            }
        }
    }

    const bool full = (row_base + 16 <= nrows);   // wave-uniform fast path

    #pragma unroll
    for (int c = 0; c < NCT; ++c) {
        const int col0 = c * 16;
        v8f acc = {0.f, 0.f, 0.f, 0.f, 0.f, 0.f, 0.f, 0.f};
        #pragma unroll
        for (int ks = 0; ks < KS; ++ks) {
            const v8h b0 = *(const v8h*)&wt[(col0 + m) * K + ks * 32 + kb];
            const v8h b1 = *(const v8h*)&wt[(col0 + m) * K + ks * 32 + 16 + kb];
            const v16h bfrag = __builtin_shufflevector(b0, b1, 0, 1, 2, 3, 4, 5, 6, 7, 8, 9, 10, 11, 12, 13, 14, 15);
            acc = __builtin_amdgcn_wmma_f32_16x16x32_f16(
                false, afrag[ks], false, bfrag, (short)0, acc, false, false);
        }
        const int col = col0 + m;
        const float s = sArr[c], t = tArr[c];
        const long r0 = (long)row_base + hi * 8;
        if (full) {
            #pragma unroll
            for (int r = 0; r < 8; ++r) {
                float val = acc[r] * s + t;
                if constexpr (EPI != 0) val = fmaxf(val, 0.0f);
                if constexpr (EPI == 1) outH[(r0 + r) * NC + col] = (_Float16)val;
                else                    outF[(r0 + r) * NC + col] = val;
            }
        } else {
            #pragma unroll
            for (int r = 0; r < 8; ++r) {
                if (r0 + r < nrows) {
                    float val = acc[r] * s + t;
                    if constexpr (EPI != 0) val = fmaxf(val, 0.0f);
                    if constexpr (EPI == 1) outH[(r0 + r) * NC + col] = (_Float16)val;
                    else                    outF[(r0 + r) * NC + col] = val;
                }
            }
        }
    }
}

// z = (1 + eps) * h
__global__ void k_scale_init(const float* __restrict__ h, float* __restrict__ z,
                             const float* __restrict__ epsp, long n4)
{
    long i = (long)blockIdx.x * 256 + threadIdx.x;
    if (i >= n4) return;
    float a = 1.0f + epsp[0];
    f32x4 v = ((const f32x4*)h)[i];
    ((f32x4*)z)[i] = v * a;
}

// z[dst] += h[src]  (1 edge x 4 floats per thread; L2-resident fp32 atomics)
__global__ void k_edge_agg(const float* __restrict__ h, const int* __restrict__ src,
                           const int* __restrict__ dst, float* __restrict__ z, int nedges)
{
    long tid = (long)blockIdx.x * 256 + threadIdx.x;
    int e = (int)(tid >> 5);
    if (e >= nedges) return;
    int f = (int)(tid & 31) << 2;
    int s = src[e], d = dst[e];
    f32x4 val = *(const f32x4*)(h + (long)s * 128 + f);
    float* zp = z + (long)d * 128 + f;
    unsafeAtomicAdd(zp + 0, val[0]);
    unsafeAtomicAdd(zp + 1, val[1]);
    unsafeAtomicAdd(zp + 2, val[2]);
    unsafeAtomicAdd(zp + 3, val[3]);
}

// pools[batch[n], coloff:coloff+128] += h[n, :]
__global__ void k_pool_add(const float* __restrict__ h, const int* __restrict__ batch,
                           float* __restrict__ pools, int coloff, int nnodes)
{
    long tid = (long)blockIdx.x * 256 + threadIdx.x;
    int node = (int)(tid >> 5);
    if (node >= nnodes) return;
    int f = (int)(tid & 31) << 2;
    int g = batch[node];
    f32x4 val = *(const f32x4*)(h + (long)node * 128 + f);
    float* pp = pools + (long)g * 768 + coloff + f;
    unsafeAtomicAdd(pp + 0, val[0]);
    unsafeAtomicAdd(pp + 1, val[1]);
    unsafeAtomicAdd(pp + 2, val[2]);
    unsafeAtomicAdd(pp + 3, val[3]);
}

__global__ void k_zero(float* __restrict__ p, int n)
{
    int i = blockIdx.x * 256 + threadIdx.x;
    if (i < n) p[i] = 0.0f;
}

// out[g, j] = pools[g, :] @ out_w[:, j] + out_b[j]   (tiny: 1024x768x128)
__global__ void k_out_gemm(const float* __restrict__ pools, const float* __restrict__ w,
                           const float* __restrict__ b, float* __restrict__ out)
{
    int g = blockIdx.x, j = threadIdx.x;
    const float* pg = pools + (long)g * 768;
    float acc = b[j];
    for (int k = 0; k < 768; ++k) acc = fmaf(pg[k], w[k * 128 + j], acc);
    out[(long)g * 128 + j] = acc;
}

extern "C" void kernel_launch(void* const* d_in, const int* in_sizes, int n_in,
                              void* d_out, int out_size, void* d_ws, size_t ws_size,
                              hipStream_t stream)
{
    const float* x     = (const float*)d_in[0];
    const int*   edge  = (const int*)d_in[1];
    const int*   batch = (const int*)d_in[2];
    const float* in_w  = (const float*)d_in[3];
    const float* in_b  = (const float*)d_in[4];
    const float* W1    = (const float*)d_in[5];
    const float* B1    = (const float*)d_in[6];
    const float* G1    = (const float*)d_in[7];
    const float* Be1   = (const float*)d_in[8];
    const float* M1    = (const float*)d_in[9];
    const float* V1    = (const float*)d_in[10];
    const float* W2    = (const float*)d_in[11];
    const float* B2    = (const float*)d_in[12];
    const float* eps   = (const float*)d_in[13];
    const float* Gn    = (const float*)d_in[14];
    const float* Bn    = (const float*)d_in[15];
    const float* Mn    = (const float*)d_in[16];
    const float* Vn    = (const float*)d_in[17];
    const float* out_w = (const float*)d_in[18];
    const float* out_b = (const float*)d_in[19];
    float* out = (float*)d_out;

    const int N = 100000, E = 1600000;
    const long chunk = (long)N * 128 * 4;          // 51.2 MB
    char* ws = (char*)d_ws;
    float*    h     = (float*)(ws);                // [N,128] f32
    float*    z     = (float*)(ws + chunk);        // [N,128] f32
    _Float16* t1    = (_Float16*)(ws + 2 * chunk); // [N,256] f16
    float*    pools = (float*)(ws + 3 * chunk);    // [1024,768] f32

    const int gemm_blocks = (N + 127) / 128;       // 782
    const int nodef_blocks = (int)(((long)N * 32 + 255) / 256);   // node x float4
    const int edgef_blocks = (int)(((long)E * 32 + 255) / 256);   // edge x float4

    k_zero<<<(1024 * 768 + 255) / 256, 256, 0, stream>>>(pools, 1024 * 768);

    // h = x @ in_w + in_b
    gemm_wmma<128, 128, 0, false><<<gemm_blocks, 256, 0, stream>>>(
        x, in_w, in_b, nullptr, nullptr, nullptr, nullptr, h, nullptr, N);
    k_pool_add<<<nodef_blocks, 256, 0, stream>>>(h, batch, pools, 0, N);

    for (int i = 0; i < 5; ++i) {
        // z = (1+eps_i)*h ; z[dst] += h[src]
        k_scale_init<<<nodef_blocks, 256, 0, stream>>>(h, z, eps + i, (long)N * 32);
        k_edge_agg<<<edgef_blocks, 256, 0, stream>>>(h, edge, edge + E, z, E);
        // t1 = relu(bn1(z @ W1_i + B1_i))  -> f16
        gemm_wmma<128, 256, 1, false><<<gemm_blocks, 256, 0, stream>>>(
            z, W1 + (long)i * 128 * 256,
            B1 + i * 256, G1 + i * 256, Be1 + i * 256, M1 + i * 256, V1 + i * 256,
            nullptr, t1, N);
        // h = relu(bn2(t1 @ W2_i + B2_i))  -> f32
        gemm_wmma<256, 128, 2, true><<<gemm_blocks, 256, 0, stream>>>(
            t1, W2 + (long)i * 256 * 128,
            B2 + i * 128, Gn + i * 128, Bn + i * 128, Mn + i * 128, Vn + i * 128,
            h, nullptr, N);
        k_pool_add<<<nodef_blocks, 256, 0, stream>>>(h, batch, pools, 128 * (i + 1), N);
    }

    k_out_gemm<<<1024, 128, 0, stream>>>(pools, out_w, out_b, out);
}
